// SequenceDecoder_20693152432196
// MI455X (gfx1250) — matrix operand order repaired and checked by reading, not compiled, not run on hardware
//
#include <hip/hip_runtime.h>
#include <hip/hip_bf16.h>

typedef __attribute__((ext_vector_type(16))) __bf16 v16bf;
typedef __attribute__((ext_vector_type(8)))  float  v8f;
typedef __attribute__((ext_vector_type(4)))  int    v4i;

#define AS1 __attribute__((address_space(1)))
#define AS3 __attribute__((address_space(3)))

#define V_     32000
#define H_     1024
#define B_     64
#define T_     32
#define KT_    32      /* K tiles = H/32 */
#define NT_OUT 2000    /* V/16  */
#define NT_G   192     /* 3H/16 */
#define NCHUNK 32      /* argmax chunks per row */
#define CW_    1000    /* V / NCHUNK */

/* ---------------- workspace layout (bytes) ---------------- */
static const size_t OFF_WOUT  = 0;                          /* 2000*32*32*16*2 = 65,536,000 */
static const size_t OFF_WIH   = OFF_WOUT + 65536000;        /* 192*32*32*16*2 =  6,291,456 */
static const size_t OFF_WHH   = OFF_WIH  + 6291456;
static const size_t OFF_EA    = OFF_WHH  + 6291456;         /* 4*32*32*16*2 = 131,072 */
static const size_t OFF_HA    = OFF_EA   + 131072;
static const size_t OFF_HNA   = OFF_HA   + 131072;
static const size_t OFF_HCUR  = OFF_HNA  + 131072;          /* 64*1024*4 = 262,144 */
static const size_t OFF_HNEW  = OFF_HCUR + 262144;
static const size_t OFF_GI    = OFF_HNEW + 262144;          /* 64*3072*4 = 786,432 */
static const size_t OFF_GH    = OFF_GI   + 786432;
static const size_t OFF_CTRL  = OFF_GH   + 786432;          /* x, pred, fin, done */
static const size_t OFF_PARTV = OFF_CTRL + 1024;            /* 64*32 floats */
static const size_t OFF_PARTI = OFF_PARTV + 8192;           /* 64*32 ints   */
static const size_t WS_NEED   = OFF_PARTI + 8192;

/* ---------------- helpers ---------------- */
__device__ __forceinline__ unsigned short f2bf(float f) {
  unsigned int u = __float_as_uint(f);
  u += 0x7FFFu + ((u >> 16) & 1u);      /* round-to-nearest-even */
  return (unsigned short)(u >> 16);
}

/* CDNA5 async global->LDS copy of 16 bytes (ASYNCcnt-tracked).
   Probe-confirmed prototype: (int4 AS1*, int4 AS3*, imm offset, imm cpol). */
__device__ __forceinline__ void async_copy16(const void* g, void* l) {
#if defined(__has_builtin) && __has_builtin(__builtin_amdgcn_global_load_async_to_lds_b128)
  __builtin_amdgcn_global_load_async_to_lds_b128((AS1 v4i*)g, (AS3 v4i*)l, 0, 0);
#else
  unsigned ldsa = (unsigned)(unsigned long long)(l);     /* low 32 bits of generic = LDS offset */
  unsigned long long ga = (unsigned long long)(g);
  asm volatile("global_load_async_to_lds_b128 %0, %1, off" :: "v"(ldsa), "v"(ga) : "memory");
#endif
}
__device__ __forceinline__ void async_wait0() {
#if defined(__has_builtin) && __has_builtin(__builtin_amdgcn_s_wait_asynccnt)
  __builtin_amdgcn_s_wait_asynccnt(0);
#else
  asm volatile("s_wait_asynccnt 0" ::: "memory");
#endif
}

/* A-fragment (16x32 bf16) index for element (row b, k) in [mtile][kt][lane][slot] order.
   ISA 05_wmma: lanes 0-15 hold K {0..7,16..23}, lanes 16-31 hold K {8..15,24..31}. */
__device__ __forceinline__ size_t a_idx(int b, int k) {
  int mtile = b >> 4, mrow = b & 15;
  int kt = k >> 5, kin = k & 31;
  int lane, slot;
  if      (kin <  8) { lane = mrow;      slot = kin;      }
  else if (kin < 16) { lane = mrow + 16; slot = kin - 8;  }
  else if (kin < 24) { lane = mrow;      slot = kin - 8;  }
  else               { lane = mrow + 16; slot = kin - 16; }
  return ((size_t)(mtile * KT_ + kt) * 32 + lane) * 16 + slot;
}

/* ---------------- prep: swizzle W[n][k] (row-major f32) into bf16 B-fragments ----------------
   B-frag layout [ntile][kt][lane][slot]: col n = ntile*16 + lane%16, k = kt*32 + (slot&7) + 16*(slot>=8) + 8*(lane>=16) */
__global__ void pack_w(const float* __restrict__ W, unsigned short* __restrict__ out, int nElems) {
  int tid = blockIdx.x * blockDim.x + threadIdx.x;
  if (tid >= nElems) return;
  int slot  =  tid        & 15;
  int lane  = (tid >>  4) & 31;
  int kt    = (tid >>  9) & 31;
  int ntile =  tid >> 14;
  int n = ntile * 16 + (lane & 15);
  int k = kt * 32 + (slot & 7) + ((slot >> 3) << 4) + ((lane >> 4) << 3);
  out[tid] = f2bf(W[(size_t)n * H_ + k]);
}

/* ---------------- init carry state ---------------- */
__global__ void init_state(const float* __restrict__ h0, const int* __restrict__ x0,
                           float* __restrict__ h_cur, int* __restrict__ x,
                           int* __restrict__ fin, int* __restrict__ done) {
  int tid = blockIdx.x * blockDim.x + threadIdx.x;
  if (tid < B_ * H_) h_cur[tid] = h0[tid];
  if (tid < B_) { x[tid] = x0[tid]; fin[tid] = 0; }
  if (tid == 0) *done = 0;
}

/* ---------------- per step: e = relu(emb[x]) packed; h packed ---------------- */
__global__ void embed_pack(const float* __restrict__ emb, const int* __restrict__ x,
                           const float* __restrict__ h_cur,
                           unsigned short* __restrict__ eA, unsigned short* __restrict__ hA) {
  int tid = blockIdx.x * blockDim.x + threadIdx.x;   /* 131072 threads */
  int which = tid >> 16;
  int idx = tid & 65535;
  int b = idx >> 10, j = idx & 1023;
  size_t di = a_idx(b, j);
  if (which == 0) {
    float e = emb[(size_t)x[b] * H_ + j];
    eA[di] = f2bf(e > 0.f ? e : 0.f);
  } else {
    hA[di] = f2bf(h_cur[idx]);
  }
}

/* ---------------- gates GEMM: gi = e@w_ih^T, gh = h@w_hh^T (raw, biases later) ----------------
   Block-uniform A source (gi blocks then gh blocks); A staged in LDS via async copy. */
__global__ void gates_gemm(const unsigned short* __restrict__ wihB,
                           const unsigned short* __restrict__ whhB,
                           const unsigned short* __restrict__ eA,
                           const unsigned short* __restrict__ hA,
                           float* __restrict__ gi, float* __restrict__ gh) {
  __shared__ unsigned short sA[4 * KT_ * 32 * 16];   /* 128 KB */
  int lane = threadIdx.x & 31;
  int job  = blockIdx.x * (blockDim.x >> 5) + (threadIdx.x >> 5);  /* 0..1535 */
  int which = (job >= 4 * NT_G) ? 1 : 0;             /* block-uniform (768 jobs = 96 blocks each) */
  int j2 = which ? job - 4 * NT_G : job;
  int ntile = j2 >> 2;
  int mtile = j2 & 3;
  const unsigned short* Asrc = which ? hA : eA;
  const v16bf* Bb = (const v16bf*)(which ? whhB : wihB);
  float* out = which ? gh : gi;

  /* stage all A-fragments (128 KB) into LDS */
  for (int r = threadIdx.x; r < 8192; r += 256)
    async_copy16((const char*)Asrc + (size_t)r * 16, (char*)sA + (size_t)r * 16);
  async_wait0();
  __syncthreads();

  v8f acc = {};
  for (int kt = 0; kt < KT_; ++kt) {
    v16bf afrag = *(const v16bf*)&sA[(size_t)((mtile * KT_ + kt) * 32 + lane) * 16];
    v16bf bfrag = Bb[((size_t)ntile * KT_ + kt) * 32 + lane];
    acc = __builtin_amdgcn_wmma_f32_16x16x32_bf16(false, afrag, false, bfrag,
                                                  (short)0, acc, false, false);
  }
  int n = ntile * 16 + (lane & 15);
  int mhi = (lane >> 4) << 3;
#pragma unroll
  for (int r = 0; r < 8; ++r) {
    int brow = mtile * 16 + r + mhi;
    out[(size_t)brow * (3 * H_) + n] = acc[r];
  }
}

/* ---------------- GRU pointwise combine; emits h_new f32 + packed bf16 A-frags ---------------- */
__global__ void gru_combine(const float* __restrict__ gi, const float* __restrict__ gh,
                            const float* __restrict__ b_ih, const float* __restrict__ b_hh,
                            const float* __restrict__ h_cur,
                            float* __restrict__ h_new, unsigned short* __restrict__ hnA) {
  int tid = blockIdx.x * blockDim.x + threadIdx.x;   /* 65536 */
  int b = tid >> 10, j = tid & 1023;
  size_t g0 = (size_t)b * (3 * H_);
  float i_r = gi[g0 + j]        + b_ih[j];
  float i_z = gi[g0 + H_ + j]   + b_ih[H_ + j];
  float i_n = gi[g0 + 2*H_ + j] + b_ih[2*H_ + j];
  float h_r = gh[g0 + j]        + b_hh[j];
  float h_z = gh[g0 + H_ + j]   + b_hh[H_ + j];
  float h_n = gh[g0 + 2*H_ + j] + b_hh[2*H_ + j];
  float r = 1.f / (1.f + expf(-(i_r + h_r)));
  float z = 1.f / (1.f + expf(-(i_z + h_z)));
  float n = tanhf(i_n + r * h_n);
  float hv = (1.f - z) * n + z * h_cur[tid];
  h_new[tid] = hv;
  hnA[a_idx(b, j)] = f2bf(hv);
}

/* ---------------- vocab projection: y = h_new @ w_out^T + b_out; write masked dec_output ----
   One wave per 16-wide N tile, all 4 M tiles (M=64). A-fragments staged in LDS (shared by the
   block's 8 waves) via CDNA5 async global->LDS loads; B streams from L2-resident bf16 weights. */
__global__ void proj_gemm(const unsigned short* __restrict__ wB,
                          const unsigned short* __restrict__ hnA,
                          const float* __restrict__ b_out,
                          const int* __restrict__ done,
                          float* __restrict__ out_t) {
  __shared__ unsigned short sA[4 * KT_ * 32 * 16];   /* 128 KB */
  int lane  = threadIdx.x & 31;
  int ntile = blockIdx.x * (blockDim.x >> 5) + (threadIdx.x >> 5); /* 0..1999 */
  const v16bf* Bb = (const v16bf*)wB;

  for (int r = threadIdx.x; r < 8192; r += 256)
    async_copy16((const char*)hnA + (size_t)r * 16, (char*)sA + (size_t)r * 16);
  async_wait0();
  __syncthreads();

  v8f acc0 = {}, acc1 = {}, acc2 = {}, acc3 = {};
  for (int kt = 0; kt < KT_; ++kt) {
    v16bf bfrag = Bb[((size_t)ntile * KT_ + kt) * 32 + lane];
    v16bf a0 = *(const v16bf*)&sA[(size_t)((0 * KT_ + kt) * 32 + lane) * 16];
    v16bf a1 = *(const v16bf*)&sA[(size_t)((1 * KT_ + kt) * 32 + lane) * 16];
    v16bf a2 = *(const v16bf*)&sA[(size_t)((2 * KT_ + kt) * 32 + lane) * 16];
    v16bf a3 = *(const v16bf*)&sA[(size_t)((3 * KT_ + kt) * 32 + lane) * 16];
    acc0 = __builtin_amdgcn_wmma_f32_16x16x32_bf16(false, a0, false, bfrag, (short)0, acc0, false, false);
    acc1 = __builtin_amdgcn_wmma_f32_16x16x32_bf16(false, a1, false, bfrag, (short)0, acc1, false, false);
    acc2 = __builtin_amdgcn_wmma_f32_16x16x32_bf16(false, a2, false, bfrag, (short)0, acc2, false, false);
    acc3 = __builtin_amdgcn_wmma_f32_16x16x32_bf16(false, a3, false, bfrag, (short)0, acc3, false, false);
  }
  int n = ntile * 16 + (lane & 15);
  float bias = b_out[n];
  int dflag = *done;
  float padv = (n == 0) ? 1.f : 0.f;
  int mhi = (lane >> 4) << 3;
#pragma unroll
  for (int r = 0; r < 8; ++r) {
    int m = r + mhi;
    out_t[(size_t)(m)      * V_ + n] = dflag ? padv : (acc0[r] + bias);
    out_t[(size_t)(16 + m) * V_ + n] = dflag ? padv : (acc1[r] + bias);
    out_t[(size_t)(32 + m) * V_ + n] = dflag ? padv : (acc2[r] + bias);
    out_t[(size_t)(48 + m) * V_ + n] = dflag ? padv : (acc3[r] + bias);
  }
}

/* ---------------- parallel partial argmax: (row, chunk) per block, first-max tie-break ---------------- */
__global__ void argmax_part(const float* __restrict__ y,
                            float* __restrict__ pVal, int* __restrict__ pIdx) {
  __shared__ float sv[256];
  __shared__ int   si[256];
  int row = blockIdx.x >> 5;        /* 64 rows */
  int chunk = blockIdx.x & 31;      /* 32 chunks of 1000 cols */
  int t = threadIdx.x;
  float bv = -3.0e38f; int bi = 0x7FFFFFFF;
  for (int v = t; v < CW_; v += 256) {
    int col = chunk * CW_ + v;
    float val = y[(size_t)row * V_ + col];
    if (val > bv || (val == bv && col < bi)) { bv = val; bi = col; }
  }
  sv[t] = bv; si[t] = bi;
  __syncthreads();
  for (int s = 128; s > 0; s >>= 1) {
    if (t < s) {
      float ov = sv[t + s]; int oi = si[t + s];
      if (ov > sv[t] || (ov == sv[t] && oi < si[t])) { sv[t] = ov; si[t] = oi; }
    }
    __syncthreads();
  }
  if (t == 0) { pVal[blockIdx.x] = sv[0]; pIdx[blockIdx.x] = si[0]; }
}

/* ---------------- final reduce + EOS/done control + hidden output + carries ---------------- */
__global__ void finalize(const float* __restrict__ pVal, const int* __restrict__ pIdx,
                         const float* __restrict__ h_new,
                         float* __restrict__ h_cur, float* __restrict__ hid_t,
                         int* __restrict__ x, int* __restrict__ pred,
                         int* __restrict__ fin, int* __restrict__ done) {
  __shared__ int s_pred[B_];
  int tid = threadIdx.x;            /* 1024 threads = 32 waves */
  int olddone = *done;              /* read before write below; barrier orders it */
  int wave = tid >> 5, lane = tid & 31;
  for (int rr = 0; rr < 2; ++rr) {
    int row = wave * 2 + rr;
    float bv = pVal[row * NCHUNK + lane];
    int   bi = pIdx[row * NCHUNK + lane];
    for (int off = 16; off > 0; off >>= 1) {
      float ov = __shfl_down(bv, off, 32);
      int   oi = __shfl_down(bi, off, 32);
      if (ov > bv || (ov == bv && oi < bi)) { bv = ov; bi = oi; }
    }
    if (lane == 0) s_pred[row] = bi;
  }
  __syncthreads();
  if (tid == 0) {
    int all = 1;
    for (int b = 0; b < B_; ++b) {
      int p = s_pred[b];
      pred[b] = p;
      int f = fin[b] | (olddone ? 0 : (p == 2));   /* EOS_IDX == 2 */
      fin[b] = f;
      all &= (f != 0);
      if (!olddone) x[b] = p;
    }
    *done = olddone | all;
  }
  for (int i = tid; i < B_ * H_; i += 1024) {
    float hv = h_new[i];
    hid_t[i] = olddone ? 0.f : hv;
    if (!olddone) h_cur[i] = hv;
  }
}

/* ---------------- driver ---------------- */
extern "C" void kernel_launch(void* const* d_in, const int* in_sizes, int n_in,
                              void* d_out, int out_size, void* d_ws, size_t ws_size,
                              hipStream_t stream) {
  (void)in_sizes; (void)n_in; (void)out_size;
  if (ws_size < WS_NEED) return;

  const int*   x0    = (const int*)  d_in[0];
  const float* h0    = (const float*)d_in[1];
  const float* emb   = (const float*)d_in[2];
  const float* w_ih  = (const float*)d_in[3];
  const float* w_hh  = (const float*)d_in[4];
  const float* b_ih  = (const float*)d_in[5];
  const float* b_hh  = (const float*)d_in[6];
  const float* w_out = (const float*)d_in[7];
  const float* b_out = (const float*)d_in[8];

  char* ws = (char*)d_ws;
  unsigned short* wOutB = (unsigned short*)(ws + OFF_WOUT);
  unsigned short* wIhB  = (unsigned short*)(ws + OFF_WIH);
  unsigned short* wHhB  = (unsigned short*)(ws + OFF_WHH);
  unsigned short* eA    = (unsigned short*)(ws + OFF_EA);
  unsigned short* hA    = (unsigned short*)(ws + OFF_HA);
  unsigned short* hnA   = (unsigned short*)(ws + OFF_HNA);
  float* hCur = (float*)(ws + OFF_HCUR);
  float* hNew = (float*)(ws + OFF_HNEW);
  float* gi   = (float*)(ws + OFF_GI);
  float* gh   = (float*)(ws + OFF_GH);
  int* xBuf    = (int*)(ws + OFF_CTRL);
  int* predBuf = (int*)(ws + OFF_CTRL + 256);
  int* finBuf  = (int*)(ws + OFF_CTRL + 512);
  int* doneBuf = (int*)(ws + OFF_CTRL + 768);
  float* pVal  = (float*)(ws + OFF_PARTV);
  int*   pIdx  = (int*)(ws + OFF_PARTI);

  float* outBase = (float*)d_out;
  float* hidBase = outBase + (size_t)T_ * B_ * V_;

  /* one-time (per call) weight swizzle to bf16 fragments: 65.5 MB w_out stays L2-resident */
  pack_w<<<(NT_OUT << 14) / 256, 256, 0, stream>>>(w_out, wOutB, NT_OUT << 14);
  pack_w<<<(NT_G   << 14) / 256, 256, 0, stream>>>(w_ih,  wIhB,  NT_G   << 14);
  pack_w<<<(NT_G   << 14) / 256, 256, 0, stream>>>(w_hh,  wHhB,  NT_G   << 14);
  init_state<<<256, 256, 0, stream>>>(h0, x0, hCur, xBuf, finBuf, doneBuf);

  for (int t = 0; t < T_; ++t) {
    float* out_t = outBase + (size_t)t * B_ * V_;
    float* hid_t = hidBase + (size_t)t * B_ * H_;
    embed_pack<<<512, 256, 0, stream>>>(emb, xBuf, hCur, eA, hA);
    gates_gemm<<<(2 * 4 * NT_G) / 8, 256, 0, stream>>>(wIhB, wHhB, eA, hA, gi, gh);
    gru_combine<<<256, 256, 0, stream>>>(gi, gh, b_ih, b_hh, hCur, hNew, hnA);
    proj_gemm<<<NT_OUT / 8, 256, 0, stream>>>(wOutB, hnA, b_out, doneBuf, out_t);
    argmax_part<<<B_ * NCHUNK, 256, 0, stream>>>(out_t, pVal, pIdx);
    finalize<<<1, 1024, 0, stream>>>(pVal, pIdx, hNew, hCur, hid_t, xBuf, predBuf, finBuf, doneBuf);
  }
}
